// Longformer_8950711844881
// MI455X (gfx1250) — matrix operand order repaired and checked by reading, not compiled, not run on hardware
//
#include <hip/hip_runtime.h>
#include <hip/hip_bf16.h>

// ---------------- problem constants ----------------
#define BB 2
#define NN 4096
#define DD 512
#define HH 8
#define HDIM 64
#define LL 2
#define WW 128
#define MM 2048
#define BN (BB * NN)
#define NEGV  (-1000000000.0f)
#define SOFTV (-10000.0f)

typedef __bf16 bf16;
typedef bf16  v16bf __attribute__((ext_vector_type(16)));
typedef float v8f   __attribute__((ext_vector_type(8)));

static __device__ inline v8f wmma_bf16(v16bf a, v16bf b, v8f c) {
  // D(f32 16x16) = A(bf16 16x32) * B(bf16 32x16) + C
  return __builtin_amdgcn_wmma_f32_16x16x32_bf16(false, a, false, b, (short)0, c,
                                                 false, false);
}

// ---------------- weight transpose + fp32->bf16 ----------------
// W row-major [K][C]  ->  WT row-major [C][K]
__global__ __launch_bounds__(256) void wt_kernel(const float* __restrict__ Wm,
                                                 bf16* __restrict__ WT,
                                                 int K, int C) {
  size_t i = (size_t)blockIdx.x * 256 + threadIdx.x;  // over K*C exactly
  int c = (int)(i % C);
  int k = (int)(i / C);
  WT[(size_t)c * K + k] = (bf16)Wm[i];
}

// ---------------- layernorm (row of 512) -> bf16 ----------------
__global__ __launch_bounds__(256) void ln_kernel(const float* __restrict__ x,
                                                 const float* __restrict__ g,
                                                 const float* __restrict__ bta,
                                                 bf16* __restrict__ h) {
  int row = blockIdx.x;
  int tid = threadIdx.x;
  const float* xr = x + (size_t)row * DD;
  __shared__ float red[8];
  float v0 = xr[tid], v1 = xr[tid + 256];
  float s = v0 + v1;
#pragma unroll
  for (int mk = 16; mk; mk >>= 1) s += __shfl_xor(s, mk, 32);
  if ((tid & 31) == 0) red[tid >> 5] = s;
  __syncthreads();
  float tot = 0.f;
#pragma unroll
  for (int i = 0; i < 8; ++i) tot += red[i];
  float mu = tot * (1.0f / DD);
  float d0 = v0 - mu, d1 = v1 - mu;
  s = d0 * d0 + d1 * d1;
  __syncthreads();
#pragma unroll
  for (int mk = 16; mk; mk >>= 1) s += __shfl_xor(s, mk, 32);
  if ((tid & 31) == 0) red[tid >> 5] = s;
  __syncthreads();
  tot = 0.f;
#pragma unroll
  for (int i = 0; i < 8; ++i) tot += red[i];
  float rstd = rsqrtf(tot * (1.0f / DD) + 1e-5f);
  h[(size_t)row * DD + tid]       = (bf16)(d0 * rstd * g[tid] + bta[tid]);
  h[(size_t)row * DD + tid + 256] = (bf16)(d1 * rstd * g[tid + 256] + bta[tid + 256]);
}

// ---------------- WMMA GEMM: Y[r,c] = act( A[r,:K] . WT[c,:K] + bias[c] ) ---------
// Each wave owns a 16x64 output tile: one A fragment feeds 4 B fragments
// (4x A-operand reuse, 4 WMMAs in flight per K-step to cover XDL latency).
// EPI 0: store bf16       EPI 1: exact GELU -> bf16       EPI 2: + R -> fp32
template <int EPI>
__global__ __launch_bounds__(256) void gemm_wmma(const bf16* __restrict__ A,
                                                 const bf16* __restrict__ WT,
                                                 const float* __restrict__ bias,
                                                 bf16* __restrict__ Yb,
                                                 float* __restrict__ Yf,
                                                 const float* __restrict__ R,
                                                 int K, int C) {
  int lane = threadIdx.x & 31;
  int wave = threadIdx.x >> 5;
  int half = lane >> 4;
  int l16  = lane & 15;
  int row0 = blockIdx.y * 128 + wave * 16;
  int col0 = blockIdx.x * 64;
  // A layout: lane = row, lanes 0-15 K 0..15, lanes 16-31 K 16..31
  const bf16* Abase = A + (size_t)(row0 + l16) * K + (half ? 16 : 0);
  // B layout: lane = column, lanes 0-15 K 0..15, lanes 16-31 K 16..31
  const bf16* Bbase = WT + (size_t)(col0 + l16) * K + (half ? 16 : 0);
  v8f acc[4];
  v8f zero = {0.f, 0.f, 0.f, 0.f, 0.f, 0.f, 0.f, 0.f};
#pragma unroll
  for (int t = 0; t < 4; ++t) acc[t] = zero;
#pragma unroll 2
  for (int k0 = 0; k0 < K; k0 += 32) {
    v16bf a;
    const bf16* pa = Abase + k0;
#pragma unroll
    for (int e = 0; e < 16; ++e) { a[e] = pa[e]; }
    if (k0 + 32 < K) __builtin_prefetch(pa + 32, 0, 1);
#pragma unroll
    for (int t = 0; t < 4; ++t) {
      const bf16* pb = Bbase + (size_t)t * 16 * K + k0;
      v16bf b;
#pragma unroll
      for (int e = 0; e < 16; ++e) b[e] = pb[e];
      if (k0 + 32 < K) __builtin_prefetch(pb + 32, 0, 1);
      acc[t] = wmma_bf16(a, b, acc[t]);
    }
  }
#pragma unroll
  for (int t = 0; t < 4; ++t) {
    int col = col0 + t * 16 + l16;
    float bc = bias[col];
#pragma unroll
    for (int g = 0; g < 8; ++g) {
      int rr = row0 + (half ? 8 : 0) + g;  // C/D layout: VGPR g -> row g / 8+g
      size_t idx = (size_t)rr * C + col;
      float v = acc[t][g] + bc;
      if (EPI == 0) {
        Yb[idx] = (bf16)v;
      } else if (EPI == 1) {
        float gl = 0.5f * v * (1.0f + erff(v * 0.70710678118f));
        Yb[idx] = (bf16)gl;
      } else {
        Yf[idx] = v + R[idx];
      }
    }
  }
}

// ---------------- V transpose:  V[b,n,h*64+hd] -> VT[b,h,hd,n] ----------------
__global__ __launch_bounds__(256) void vt_kernel(const bf16* __restrict__ V,
                                                 bf16* __restrict__ VT) {
  size_t i = (size_t)blockIdx.x * 256 + threadIdx.x;  // over BN*DD exactly
  int d = (int)(i % DD);
  size_t bn = i / DD;
  int n = (int)(bn % NN);
  int b = (int)(bn / NN);
  int h = d >> 6, hd = d & 63;
  VT[((size_t)(b * HH + h) * HDIM + hd) * NN + n] = V[i];
}

// ---------------- flash-style windowed attention ----------------
// grid: (N/W blocks, H, B); 8 waves, one 16-query tile each
__global__ __launch_bounds__(256) void attn_kernel(const bf16* __restrict__ Q,
                                                   const bf16* __restrict__ Kc,
                                                   const bf16* __restrict__ Vc,
                                                   const bf16* __restrict__ VT,
                                                   float* __restrict__ OUT) {
  const int c = blockIdx.x, h = blockIdx.y, b = blockIdx.z;
  const int lane = threadIdx.x & 31, wave = threadIdx.x >> 5;
  const int half = lane >> 4, l16 = lane & 15;
  const int qr0 = c * WW + wave * 16;  // absolute first query row of this wave
  const float scale = 0.125f;          // 1/sqrt(64)
  __shared__ bf16 pbuf[8][16][32];     // per-wave P tile (C-layout -> A-layout)

  const int aoff = half ? 16 : 0;
  const bf16* qp = Q + ((size_t)(b * NN + qr0 + l16) * DD + h * HDIM);
  v16bf aq0, aq1;  // q A-fragments for K=0..31 and 32..63
#pragma unroll
  for (int e = 0; e < 16; ++e) {
    aq0[e] = qp[aoff + e];
    aq1[e] = qp[32 + aoff + e];
  }

  float mrow[8], lrow[8];
  v8f zero = {0.f, 0.f, 0.f, 0.f, 0.f, 0.f, 0.f, 0.f};
  v8f o[4];
#pragma unroll
  for (int g = 0; g < 8; ++g) { mrow[g] = -1e30f; lrow[g] = 0.f; }
#pragma unroll
  for (int t = 0; t < 4; ++t) o[t] = zero;

  const int kw0 = c * WW - WW;  // window start (absolute key pos), 3W keys total
  for (int j = 0; j < (3 * WW) / 32; ++j) {  // 12 chunks of 32 keys
    float sv0[8], sv1[8];
#pragma unroll
    for (int s = 0; s < 2; ++s) {
      int ky = kw0 + j * 32 + s * 16 + l16;  // this lane's key column
      bool okcol = (ky >= 0) && (ky < NN);
      int kyc = okcol ? ky : 0;
      const bf16* kp = Kc + ((size_t)(b * NN + kyc) * DD + h * HDIM) + (half ? 16 : 0);
      v16bf bk0, bk1;
#pragma unroll
      for (int e = 0; e < 16; ++e) { bk0[e] = kp[e]; bk1[e] = kp[32 + e]; }
      v8f cs = zero;
      cs = wmma_bf16(aq0, bk0, cs);
      cs = wmma_bf16(aq1, bk1, cs);
      float* dst = s ? sv1 : sv0;
#pragma unroll
      for (int g = 0; g < 8; ++g) {
        int rr = qr0 + (half ? 8 : 0) + g;   // absolute query position
        int rel = ky - rr;
        float vvv = cs[g] * scale;
        bool ok = okcol && (rel <= WW) && (rel >= -WW);
        vvv = ok ? vvv : NEGV;
        if (ky == 0) vvv += SOFTV;
        dst[g] = vvv;
      }
    }
    // online softmax update over this 32-key chunk (per-row, 16-lane butterflies)
#pragma unroll
    for (int g = 0; g < 8; ++g) {
      float mx = fmaxf(sv0[g], sv1[g]);
#pragma unroll
      for (int mk = 1; mk < 16; mk <<= 1) mx = fmaxf(mx, __shfl_xor(mx, mk, 32));
      float mn = fmaxf(mrow[g], mx);
      float alpha = __expf(mrow[g] - mn);
      mrow[g] = mn;
      float p0 = __expf(sv0[g] - mn);
      float p1 = __expf(sv1[g] - mn);
      float rs = p0 + p1;
#pragma unroll
      for (int mk = 1; mk < 16; mk <<= 1) rs += __shfl_xor(rs, mk, 32);
      lrow[g] = lrow[g] * alpha + rs;
#pragma unroll
      for (int t = 0; t < 4; ++t) o[t][g] *= alpha;
      int rit = (half ? 8 : 0) + g;
      pbuf[wave][rit][l16]      = (bf16)p0;
      pbuf[wave][rit][16 + l16] = (bf16)p1;
    }
    __syncthreads();
    // reload P as an A-fragment (16 rows x 32 keys)
    v16bf ap;
    const bf16* pp = &pbuf[wave][l16][0];
#pragma unroll
    for (int e = 0; e < 16; ++e) { ap[e] = pp[aoff + e]; }
    int kb = half ? 16 : 0;
#pragma unroll
    for (int t = 0; t < 4; ++t) {  // hd tiles 0..3
      int hd = t * 16 + l16;
      const bf16* vp = VT + ((size_t)(b * HH + h) * HDIM + hd) * NN;
      v16bf bvf;
#pragma unroll
      for (int e = 0; e < 16; ++e) {
        int kk = kw0 + j * 32 + kb + e;
        kk = kk < 0 ? 0 : (kk >= NN ? NN - 1 : kk);  // masked keys have P==0
        bvf[e] = vp[kk];
      }
      o[t] = wmma_bf16(ap, bvf, o[t]);
    }
    __syncthreads();
  }

  // fold in the global column: sg = scale * q . k[:,0];  out += pg * v[:,0]
  float sg = 0.f;
  {
    const bf16* qrp = Q + ((size_t)(b * NN + qr0 + l16) * DD + h * HDIM);
    const bf16* k0p = Kc + ((size_t)(b * NN) * DD + h * HDIM);
#pragma unroll 16
    for (int i = 0; i < HDIM; ++i) sg += (float)qrp[i] * (float)k0p[i];
    sg *= scale;
  }
#pragma unroll
  for (int g = 0; g < 8; ++g) {
    int rit = (half ? 8 : 0) + g;
    float sgr = __shfl(sg, rit, 32);  // lanes 0..15 hold sg for rows 0..15
    float mn = fmaxf(mrow[g], sgr);
    float alpha = __expf(mrow[g] - mn);
    float pg = __expf(sgr - mn);
    mrow[g] = mn;
    lrow[g] = lrow[g] * alpha + pg;
#pragma unroll
    for (int t = 0; t < 4; ++t) {
      int hd = t * 16 + l16;
      float v0 = (float)Vc[(size_t)(b * NN) * DD + h * HDIM + hd];
      o[t][g] = o[t][g] * alpha + pg * v0;
    }
  }
#pragma unroll
  for (int t = 0; t < 4; ++t) {
    int hd = t * 16 + l16;
#pragma unroll
    for (int g = 0; g < 8; ++g) {
      int rr = qr0 + (half ? 8 : 0) + g;
      OUT[(size_t)(b * NN + rr) * DD + h * HDIM + hd] = o[t][g] / lrow[g];
    }
  }
}

// ---------------- global token attention (overwrites OUT row 0) ----------------
// grid: B*H workgroups, 256 threads
__global__ __launch_bounds__(256) void gtok_kernel(const bf16* __restrict__ Hb,
                                                   const float* __restrict__ Wqg,
                                                   const float* __restrict__ bqg,
                                                   const bf16* __restrict__ KG,
                                                   const bf16* __restrict__ VG,
                                                   float* __restrict__ OUT) {
  int h = blockIdx.x & 7;
  int b = blockIdx.x >> 3;
  int tid = threadIdx.x;
  __shared__ float qg[HDIM];
  __shared__ float sred[256];
  __shared__ float sbuf[NN];
  if (tid < HDIM) {
    int col = h * HDIM + tid;
    float acc = bqg[col];
    const bf16* hr = Hb + (size_t)(b * NN) * DD;  // row 0 of batch b
    for (int k = 0; k < DD; ++k) acc += (float)hr[k] * Wqg[(size_t)k * DD + col];
    qg[tid] = acc * 0.125f;
  }
  __syncthreads();
  for (int n = tid; n < NN; n += 256) {
    const bf16* kp = KG + ((size_t)(b * NN + n) * DD + h * HDIM);
    float s = 0.f;
#pragma unroll 16
    for (int i = 0; i < HDIM; ++i) s += qg[i] * (float)kp[i];
    sbuf[n] = s;
  }
  __syncthreads();
  float mx = -1e30f;
  for (int n = tid; n < NN; n += 256) mx = fmaxf(mx, sbuf[n]);
  sred[tid] = mx;
  __syncthreads();
  for (int s2 = 128; s2; s2 >>= 1) {
    if (tid < s2) sred[tid] = fmaxf(sred[tid], sred[tid + s2]);
    __syncthreads();
  }
  float m = sred[0];
  __syncthreads();
  float ls = 0.f;
  for (int n = tid; n < NN; n += 256) {
    float p = __expf(sbuf[n] - m);
    sbuf[n] = p;
    ls += p;
  }
  sred[tid] = ls;
  __syncthreads();
  for (int s2 = 128; s2; s2 >>= 1) {
    if (tid < s2) sred[tid] += sred[tid + s2];
    __syncthreads();
  }
  float denom = sred[0];
  __syncthreads();
  int hd = tid & 63, part = tid >> 6;  // 4 partial sums per hd
  float acc = 0.f;
  for (int n = part * (NN / 4); n < (part + 1) * (NN / 4); ++n)
    acc += sbuf[n] * (float)VG[(size_t)(b * NN + n) * DD + h * HDIM + hd];
  sred[tid] = acc;
  __syncthreads();
  if (part == 0) {
    float tot = sred[hd] + sred[hd + 64] + sred[hd + 128] + sred[hd + 192];
    OUT[(size_t)(b * NN) * DD + h * HDIM + hd] = tot / denom;
  }
}

// ---------------- residual add ----------------
__global__ __launch_bounds__(256) void add_kernel(float* __restrict__ x,
                                                  const float* __restrict__ a) {
  size_t i = (size_t)blockIdx.x * 256 + threadIdx.x;
  x[i] += a[i];
}

// ---------------- host orchestration ----------------
extern "C" void kernel_launch(void* const* d_in, const int* in_sizes, int n_in,
                              void* d_out, int out_size, void* d_ws, size_t ws_size,
                              hipStream_t stream) {
  (void)in_sizes; (void)n_in; (void)out_size; (void)ws_size;
  const float* x    = (const float*)d_in[0];
  const float* ln1g = (const float*)d_in[1];
  const float* ln1b = (const float*)d_in[2];
  const float* Wq   = (const float*)d_in[3];
  const float* bq   = (const float*)d_in[4];
  const float* Wk   = (const float*)d_in[5];
  const float* bk   = (const float*)d_in[6];
  const float* Wv   = (const float*)d_in[7];
  const float* bv   = (const float*)d_in[8];
  const float* Wqg  = (const float*)d_in[9];
  const float* bqg  = (const float*)d_in[10];
  const float* Wkg  = (const float*)d_in[11];
  const float* bkg  = (const float*)d_in[12];
  const float* Wvg  = (const float*)d_in[13];
  const float* bvg  = (const float*)d_in[14];
  const float* ln2g = (const float*)d_in[15];
  const float* ln2b = (const float*)d_in[16];
  const float* W1   = (const float*)d_in[17];
  const float* b1   = (const float*)d_in[18];
  const float* W2   = (const float*)d_in[19];
  const float* b2   = (const float*)d_in[20];
  float* xout = (float*)d_out;
  char* ws = (char*)d_ws;

  size_t off = 0;
  auto alloc = [&](size_t bytes) {
    size_t r = off;
    off += (bytes + 255) & ~(size_t)255;
    return r;
  };
  const size_t szDD_bf  = (size_t)DD * DD * sizeof(bf16);
  const size_t szDM_bf  = (size_t)DD * MM * sizeof(bf16);
  const size_t szAct_bf = (size_t)BN * DD * sizeof(bf16);

  bf16* WqT  = (bf16*)(ws + alloc(szDD_bf));
  bf16* WkT  = (bf16*)(ws + alloc(szDD_bf));
  bf16* WvT  = (bf16*)(ws + alloc(szDD_bf));
  bf16* WkgT = (bf16*)(ws + alloc(szDD_bf));
  bf16* WvgT = (bf16*)(ws + alloc(szDD_bf));
  bf16* W1T  = (bf16*)(ws + alloc(szDM_bf));
  bf16* W2T  = (bf16*)(ws + alloc(szDM_bf));
  bf16* Hb   = (bf16*)(ws + alloc(szAct_bf));
  size_t qoff = alloc(szAct_bf);
  bf16* Qb   = (bf16*)(ws + qoff);
  bf16* Kb   = (bf16*)(ws + alloc(szAct_bf));
  bf16* Vb   = (bf16*)(ws + alloc(szAct_bf));
  bf16* KGb  = (bf16*)(ws + alloc(szAct_bf));
  bf16* VGb  = (bf16*)(ws + alloc(szAct_bf));
  bf16* VTb  = (bf16*)(ws + alloc(szAct_bf));
  float* ATT = (float*)(ws + alloc((size_t)BN * DD * sizeof(float)));
  // FFN hidden [BN x M] bf16 = 32MB aliases Qb..KGb (4 x 8MB, dead by FFN time)
  bf16* HID  = (bf16*)(ws + qoff);

  hipMemcpyAsync(xout, x, (size_t)BN * DD * sizeof(float),
                 hipMemcpyDeviceToDevice, stream);

  dim3 blk(256);
  dim3 gProj(DD / 64, BN / 128);
  dim3 gFfn1(MM / 64, BN / 128);
  dim3 gFfn2(DD / 64, BN / 128);
  int nElemAct = (BN * DD) / 256;

  for (int l = 0; l < LL; ++l) {
    const float* Wq_l  = Wq  + (size_t)l * DD * DD;
    const float* Wk_l  = Wk  + (size_t)l * DD * DD;
    const float* Wv_l  = Wv  + (size_t)l * DD * DD;
    const float* Wkg_l = Wkg + (size_t)l * DD * DD;
    const float* Wvg_l = Wvg + (size_t)l * DD * DD;
    const float* Wqg_l = Wqg + (size_t)l * DD * DD;
    const float* W1_l  = W1  + (size_t)l * DD * MM;
    const float* W2_l  = W2  + (size_t)l * MM * DD;

    int nDD = (DD * DD) / 256;
    wt_kernel<<<nDD, blk, 0, stream>>>(Wq_l,  WqT,  DD, DD);
    wt_kernel<<<nDD, blk, 0, stream>>>(Wk_l,  WkT,  DD, DD);
    wt_kernel<<<nDD, blk, 0, stream>>>(Wv_l,  WvT,  DD, DD);
    wt_kernel<<<nDD, blk, 0, stream>>>(Wkg_l, WkgT, DD, DD);
    wt_kernel<<<nDD, blk, 0, stream>>>(Wvg_l, WvgT, DD, DD);
    wt_kernel<<<(DD * MM) / 256, blk, 0, stream>>>(W1_l, W1T, DD, MM);
    wt_kernel<<<(MM * DD) / 256, blk, 0, stream>>>(W2_l, W2T, MM, DD);

    ln_kernel<<<BN, blk, 0, stream>>>(xout, ln1g + l * DD, ln1b + l * DD, Hb);

    gemm_wmma<0><<<gProj, blk, 0, stream>>>(Hb, WqT,  bq  + l * DD, Qb,  nullptr, nullptr, DD, DD);
    gemm_wmma<0><<<gProj, blk, 0, stream>>>(Hb, WkT,  bk  + l * DD, Kb,  nullptr, nullptr, DD, DD);
    gemm_wmma<0><<<gProj, blk, 0, stream>>>(Hb, WvT,  bv  + l * DD, Vb,  nullptr, nullptr, DD, DD);
    gemm_wmma<0><<<gProj, blk, 0, stream>>>(Hb, WkgT, bkg + l * DD, KGb, nullptr, nullptr, DD, DD);
    gemm_wmma<0><<<gProj, blk, 0, stream>>>(Hb, WvgT, bvg + l * DD, VGb, nullptr, nullptr, DD, DD);

    vt_kernel<<<nElemAct, blk, 0, stream>>>(Vb, VTb);
    attn_kernel<<<dim3(NN / WW, HH, BB), blk, 0, stream>>>(Qb, Kb, Vb, VTb, ATT);
    gtok_kernel<<<BB * HH, blk, 0, stream>>>(Hb, Wqg_l, bqg + l * DD, KGb, VGb, ATT);
    add_kernel<<<nElemAct, blk, 0, stream>>>(xout, ATT);

    ln_kernel<<<BN, blk, 0, stream>>>(xout, ln2g + l * DD, ln2b + l * DD, Hb);
    gemm_wmma<1><<<gFfn1, blk, 0, stream>>>(Hb, W1T, b1 + l * MM, HID, nullptr, nullptr, DD, MM);
    gemm_wmma<2><<<gFfn2, blk, 0, stream>>>(HID, W2T, b2 + l * DD, nullptr, xout, xout, MM, DD);
  }
}
